// QuantizedLinear_68393059222044
// MI455X (gfx1250) — matrix operand order repaired and checked by reading, not compiled, not run on hardware
//
#include <hip/hip_runtime.h>

// ---------------------------------------------------------------------------
// W4A32 grouped-quant linear:  out[M,N] = x[M,K] * dequant(qw)[N,K]^T
//   M = 8192 (B*S), N = 4096 (OUT), K = 4096 (IN), group = 128, 4-bit packed
//   (one byte = two nibbles per int32 element of qweight).
//
// MI455X strategy:
//  * f16 WMMA GEMM (v_wmma_f32_16x16x32_f16, f32 accum) — compute-bound at
//    ~275 GFLOP; HBM floor ~290MB (~12.5us), so feed the WMMA pipes.
//  * dequant-on-the-fly: x (128MB) + qweight (32MB) stay resident in 192MB L2
//    across the 32 N-tile passes; output uses non-temporal stores so the
//    128MB write-once stream doesn't evict them.
//  * double-buffered LDS tiles + register prefetch: global loads for tile k+1
//    issue before the WMMA chain of tile k; one barrier per K-step.
// ---------------------------------------------------------------------------

typedef _Float16 v8h   __attribute__((ext_vector_type(8)));
typedef _Float16 v16h  __attribute__((ext_vector_type(16)));
typedef float    v8f   __attribute__((ext_vector_type(8)));
typedef float    f32x4 __attribute__((ext_vector_type(4)));
typedef int      i32x4 __attribute__((ext_vector_type(4)));

#define BM 128
#define BN 128
#define BK 32
#define LDSTR 40   // f16 per row (32 + 8 pad); 80B row stride keeps 16B align

// Convert prefetched registers into the LDS tile buffers.
__device__ __forceinline__ void stage_tiles(_Float16* __restrict__ Asb,
                                            _Float16* __restrict__ Bsb,
                                            int row, int half,
                                            f32x4 f0, f32x4 f1, f32x4 f2, f32x4 f3,
                                            i32x4 q0, i32x4 q1, float s, float z)
{
    v8h a0, a1;
    #pragma unroll
    for (int u = 0; u < 4; ++u) {
        a0[u]     = (_Float16)f0[u];
        a0[4 + u] = (_Float16)f1[u];
        a1[u]     = (_Float16)f2[u];
        a1[4 + u] = (_Float16)f3[u];
    }
    _Float16* ad = Asb + row * LDSTR + half * 16;
    *(v8h*)(ad + 0) = a0;
    *(v8h*)(ad + 8) = a1;

    v8h b0, b1;
    #pragma unroll
    for (int u = 0; u < 4; ++u) {
        int v = q0[u];
        b0[2 * u]     = (_Float16)(((float)( v       & 15) - z) * s);
        b0[2 * u + 1] = (_Float16)(((float)((v >> 4) & 15) - z) * s);
        v = q1[u];
        b1[2 * u]     = (_Float16)(((float)( v       & 15) - z) * s);
        b1[2 * u + 1] = (_Float16)(((float)((v >> 4) & 15) - z) * s);
    }
    _Float16* bd = Bsb + row * LDSTR + half * 16;
    *(v8h*)(bd + 0) = b0;
    *(v8h*)(bd + 8) = b1;
}

__global__ __launch_bounds__(256, 2)
void qlinear_wmma_f16(const float* __restrict__ x,
                      const int*   __restrict__ qw,
                      const float* __restrict__ scale,
                      const float* __restrict__ zero,
                      float*       __restrict__ out,
                      int M, int N, int K)
{
    __shared__ __align__(16) _Float16 As[2][BM * LDSTR];  // [m][k]
    __shared__ __align__(16) _Float16 Bs[2][BN * LDSTR];  // [n][k] (= W row-major)

    const int tid  = threadIdx.x;
    const int lane = tid & 31;
    const int wave = tid >> 5;        // 0..7
    const int wm   = wave >> 1;       // 0..3 : 32-row strip
    const int wn   = wave & 1;        // 0..1 : 64-col strip

    // cooperative loaders: 2 threads per row, each covers 16 K-values
    const int row  = tid >> 1;        // 0..127
    const int half = tid & 1;         // 0 or 1

    const long mBase = (long)blockIdx.y * BM;
    const long nBase = (long)blockIdx.x * BN;

    const int laneM  = lane & 15;
    const int aKlo   = (lane < 16) ? 0 : 8;   // A frag: K = aKlo..+7 and aKlo+16..+23
    const int bKbase = (lane < 16) ? 0 : 8;   // B frag: K = bKbase..+7 and bKbase+16..+23

    const int  kTiles  = K / BK;              // 128
    const int  nGroups = K >> 7;              // K / 128

    const float* xPtr = x  + (mBase + row) * (long)K        + half * 16;
    const int*   qPtr = qw + (nBase + row) * (long)(K >> 1) + half * 8;
    const float* sPtr = scale + (nBase + row) * (long)nGroups;
    const float* zPtr = zero  + (nBase + row) * (long)nGroups;

    v8f acc[2][4];
    #pragma unroll
    for (int i = 0; i < 2; ++i)
        #pragma unroll
        for (int j = 0; j < 4; ++j)
            acc[i][j] = v8f{};

    // ---- prologue: stage tile 0 into buffer 0 ---------------------------
    {
        f32x4 f0 = *(const f32x4*)(xPtr + 0);
        f32x4 f1 = *(const f32x4*)(xPtr + 4);
        f32x4 f2 = *(const f32x4*)(xPtr + 8);
        f32x4 f3 = *(const f32x4*)(xPtr + 12);
        i32x4 q0 = *(const i32x4*)(qPtr + 0);
        i32x4 q1 = *(const i32x4*)(qPtr + 4);
        float s = sPtr[0], z = zPtr[0];
        stage_tiles(As[0], Bs[0], row, half, f0, f1, f2, f3, q0, q1, s, z);
    }
    __syncthreads();

    for (int kb = 0; kb < kTiles; ++kb) {
        const int  cur     = kb & 1;
        const int  nxt     = cur ^ 1;
        const bool hasNext = (kb + 1) < kTiles;

        // ---- prefetch tile kb+1 from global into registers --------------
        f32x4 f0{}, f1{}, f2{}, f3{};
        i32x4 q0{}, q1{};
        float s = 0.f, z = 0.f;
        if (hasNext) {
            const float* xs = xPtr + (long)(kb + 1) * BK;
            f0 = *(const f32x4*)(xs + 0);
            f1 = *(const f32x4*)(xs + 4);
            f2 = *(const f32x4*)(xs + 8);
            f3 = *(const f32x4*)(xs + 12);
            const int* qs = qPtr + (long)(kb + 1) * (BK >> 1);
            q0 = *(const i32x4*)(qs + 0);
            q1 = *(const i32x4*)(qs + 4);
            const int g = (kb + 1) >> 2;      // 128-wide quant group
            s = sPtr[g];
            z = zPtr[g];
        }

        // ---- fragments + 8 WMMAs per wave from buffer `cur` -------------
        v16h a[2], b[4];
        #pragma unroll
        for (int i = 0; i < 2; ++i) {
            const _Float16* ap = &As[cur][(wm * 32 + i * 16 + laneM) * LDSTR];
            v8h lo = *(const v8h*)(ap + aKlo);
            v8h hi = *(const v8h*)(ap + aKlo + 16);
            a[i] = __builtin_shufflevector(lo, hi, 0,1,2,3,4,5,6,7,8,9,10,11,12,13,14,15);
        }
        #pragma unroll
        for (int j = 0; j < 4; ++j) {
            const _Float16* bp = &Bs[cur][(wn * 64 + j * 16 + laneM) * LDSTR + bKbase];
            v8h lo = *(const v8h*)(bp + 0);
            v8h hi = *(const v8h*)(bp + 16);
            b[j] = __builtin_shufflevector(lo, hi, 0,1,2,3,4,5,6,7,8,9,10,11,12,13,14,15);
        }
        #pragma unroll
        for (int i = 0; i < 2; ++i)
            #pragma unroll
            for (int j = 0; j < 4; ++j)
                acc[i][j] = __builtin_amdgcn_wmma_f32_16x16x32_f16(
                    false, a[i], false, b[j], (short)0, acc[i][j], false, false);

        // ---- convert + store prefetched tile into buffer `nxt` ----------
        if (hasNext)
            stage_tiles(As[nxt], Bs[nxt], row, half, f0, f1, f2, f3, q0, q1, s, z);

        __syncthreads();   // one barrier per K-step (double buffer)
    }

    // ---- epilogue: C layout — VGPR r -> M = r (+8 for lanes 16..31), lane -> N
    // Non-temporal stores: keep x/qweight L2-resident across N-tile passes.
    const long mW   = mBase + wm * 32;
    const long nW   = nBase + wn * 64;
    const int  mOff = (lane < 16) ? 0 : 8;
    #pragma unroll
    for (int i = 0; i < 2; ++i) {
        #pragma unroll
        for (int j = 0; j < 4; ++j) {
            float* op = out + (mW + i * 16 + mOff) * (long)N + (nW + j * 16 + laneM);
            #pragma unroll
            for (int r = 0; r < 8; ++r)
                __builtin_nontemporal_store(acc[i][j][r], op + (long)r * N);
        }
    }
}

extern "C" void kernel_launch(void* const* d_in, const int* in_sizes, int n_in,
                              void* d_out, int out_size, void* d_ws, size_t ws_size,
                              hipStream_t stream)
{
    const float* x     = (const float*)d_in[0];
    const int*   qw    = (const int*)  d_in[1];
    const float* scale = (const float*)d_in[2];
    const float* zero  = (const float*)d_in[3];
    float*       out   = (float*)d_out;

    const int K = 4096;                 // IN
    const int N = 4096;                 // OUT
    const int M = in_sizes[0] / K;      // B * S = 8192

    dim3 grid(N / BN, M / BM);
    dim3 block(256);
    qlinear_wmma_f16<<<grid, block, 0, stream>>>(x, qw, scale, zero, out, M, N, K);
}
